// RNNLayer_47631187313203
// MI455X (gfx1250) — matrix operand order repaired and checked by reading, not compiled, not run on hardware
//
#include <hip/hip_runtime.h>
#include <hip/hip_bf16.h>
#include <math.h>

// Problem dims (fixed by the reference)
constexpr int Bb = 64, Tt = 512, Ii = 1024, Hh = 2048;
constexpr int BT = Bb * Tt;            // 32768 rows of x_proj
constexpr int NWG_REC = Hh / 64;       // 32 persistent workgroups in recurrence

#define USE_TDM 1                      // Tensor Data Mover for w_hh -> LDS

typedef _Float16 v4h  __attribute__((ext_vector_type(4)));
typedef _Float16 v8h  __attribute__((ext_vector_type(8)));
typedef _Float16 v16h __attribute__((ext_vector_type(16)));
typedef float    v8f  __attribute__((ext_vector_type(8)));
typedef unsigned int v4u __attribute__((ext_vector_type(4)));
typedef int      v4i  __attribute__((ext_vector_type(4)));
typedef int      v8i  __attribute__((ext_vector_type(8)));

static __device__ __forceinline__ v16h cat8(v8h lo, v8h hi) {
  return __builtin_shufflevector(lo, hi, 0,1,2,3,4,5,6,7,8,9,10,11,12,13,14,15);
}

static __device__ __forceinline__ v8f wmma_f16(v16h a, v16h b, v8f c) {
  // D = A(16x32 f16) x B(32x16 f16) + C(16x16 f32)
  return __builtin_amdgcn_wmma_f32_16x16x32_f16(false, a, false, b, (short)0, c,
                                                false, false);
}

// ---------------------------------------------------------------------------
// Kernel 0: prep — f32->f16 conversion of w_hh, h0 -> hbuf0 (f16), barrier=0
// ---------------------------------------------------------------------------
__global__ void prep_kernel(const float* __restrict__ whh,
                            const float* __restrict__ h0,
                            _Float16* __restrict__ w16,
                            _Float16* __restrict__ hbuf0,
                            unsigned* __restrict__ bar) {
  size_t i = (size_t)blockIdx.x * blockDim.x + threadIdx.x;
  if (i < (size_t)Hh * Hh) w16[i] = (_Float16)whh[i];
  if (i < (size_t)Bb * Hh) hbuf0[i] = (_Float16)h0[i];
  if (i == 0) *bar = 0u;
}

// ---------------------------------------------------------------------------
// Kernel 1: x_proj GEMM.  Out[m,n] = sum_k X[m,k] * W[n,k] + bias[n]
// M=32768, N=2048, K=1024. 128x128 macro-tile per WG, K-step 32, f16 staging
// in LDS, f32 accumulation via v_wmma_f32_16x16x32_f16.
// ---------------------------------------------------------------------------
__global__ __launch_bounds__(256) void xproj_gemm(
    const float* __restrict__ X,      // input  [BT][I]  (row-major, k-contig)
    const float* __restrict__ W,      // w_ih   [H][I]   (row-major, k-contig)
    const float* __restrict__ bias,   // b_ih   [H]
    float* __restrict__ Out)          // d_out  [BT][H]  (x_proj staging)
{
  __shared__ _Float16 lA[128 * 32];   // 8 KB
  __shared__ _Float16 lB[128 * 32];   // 8 KB

  const int m_base = blockIdx.y * 128;
  const int n_base = blockIdx.x * 128;
  const int tid  = threadIdx.x;
  const int lane = tid & 31;
  const int wv   = tid >> 5;          // wave 0..7 -> n-tile
  const int l15  = lane & 15;
  const int kh   = lane >> 4;         // k-half selector / C-row-half

  v8f acc[8] = {};                    // 8 m-tiles x (16x16 f32)

  for (int kb = 0; kb < Ii; kb += 32) {
    // Stage 128x32 tiles of A and B, converting f32 -> f16.
    #pragma unroll
    for (int r = 0; r < 4; ++r) {
      int j   = tid + 256 * r;        // quad index 0..1023
      int row = j >> 3;
      int kq  = (j & 7) << 2;
      float4 a4 = *(const float4*)(X + (size_t)(m_base + row) * Ii + kb + kq);
      float4 b4 = *(const float4*)(W + (size_t)(n_base + row) * Ii + kb + kq);
      v4h ap = {(_Float16)a4.x, (_Float16)a4.y, (_Float16)a4.z, (_Float16)a4.w};
      v4h bp = {(_Float16)b4.x, (_Float16)b4.y, (_Float16)b4.z, (_Float16)b4.w};
      *(v4h*)&lA[row * 32 + kq] = ap;
      *(v4h*)&lB[row * 32 + kq] = bp;
    }
    __syncthreads();

    // B fragment (32x16): lane = column n, 16 contiguous k at 16*kh.
    const _Float16* brow = &lB[(wv * 16 + l15) * 32];
    v16h bf = cat8(*(const v8h*)(brow + kh * 16),
                   *(const v8h*)(brow + kh * 16 + 8));
    #pragma unroll
    for (int mt = 0; mt < 8; ++mt) {
      // A fragment (16x32): lane = row m, 16 contiguous k at 16*kh.
      const _Float16* arow = &lA[(mt * 16 + l15) * 32];
      v16h af = cat8(*(const v8h*)(arow + kh * 16),
                     *(const v8h*)(arow + kh * 16 + 8));
      acc[mt] = wmma_f16(af, bf, acc[mt]);
    }
    __syncthreads();
  }

  // Epilogue: C/D layout — row = r + 8*kh, col = l15.
  const int n  = n_base + wv * 16 + l15;
  const float bn = bias[n];
  #pragma unroll
  for (int mt = 0; mt < 8; ++mt) {
    #pragma unroll
    for (int r = 0; r < 8; ++r) {
      int m = m_base + mt * 16 + r + 8 * kh;
      Out[(size_t)m * Hh + n] = acc[mt][r] + bn;
    }
  }
}

// ---------------------------------------------------------------------------
// Kernel 2: persistent recurrence. 32 WGs; WG j owns h-columns [64j, 64j+64).
// Its w_hh slice (64x2048 f16 = 256 KB) is DMA'd into LDS once by the Tensor
// Data Mover and stays for all 512 steps. Per step:
//   C(64x64) = h(64x2048) x w_slice^T  (f16 WMMA, f32 accum)
//   h_new = tanh(xp + noise + C + b_hh); f32 -> d_out (over x_proj, in
//   place), f16 -> ping-pong h buffer. Device-scope atomic barrier per step.
// ---------------------------------------------------------------------------
__global__ __launch_bounds__(256) void rnn_recur(
    const _Float16* __restrict__ w16,    // w_hh f16 [H][H] (row n, k-contig)
    const float* __restrict__ noise,     // internal [B][T]
    const float* __restrict__ bhh,       // b_hh [H]
    float* __restrict__ Out,             // [B*T*H] outs ++ [B*H] h_last
    _Float16* __restrict__ hbuf0,        // h ping
    _Float16* __restrict__ hbuf1,        // h pong
    unsigned* __restrict__ bar)          // global barrier counter
{
  __shared__ _Float16 wlds[64 * 2048];   // 256 KB (<= 320 KB CDNA5 WGP LDS)

  const int tid  = threadIdx.x;
  const int lane = tid & 31;
  const int wv   = tid >> 5;             // 0..7
  const int l15  = lane & 15;
  const int kh   = lane >> 4;
  const int n_base = blockIdx.x * 64;

#if USE_TDM
  // One-time TDM DMA: 2D tile (64 rows x 512 8-byte units) cut from the
  // 2048-row f16 weight tensor, straight into LDS. D# per ISA §8.
  // clang-23 6-arg form: (v4u g0, v8i g1, v4i g2, v4i g3, v8i g4, i32 cpol).
  if (tid < 32) {                        // one wave issues; EXEC ignored
    unsigned long long ga =
        (unsigned long long)(size_t)(const void*)(w16 + (size_t)n_base * Hh);
    unsigned int ldsa = (unsigned int)(size_t)(void*)wlds; // LDS byte offset
    v4u g0 = { 1u,                                  // count=1 (valid), user
               ldsa,                                // lds_addr
               (unsigned int)ga,                    // global_addr[31:0]
               (unsigned int)((ga >> 32) & 0x01FFFFFFu) | (2u << 30) }; // +type=2
    v8i g1 = { (int)(3u << 16),    // wg_mask=0 | data_size=3 (8B units)
               (int)(512u << 16),  // abar_addr=0 | tensor_dim0=512 (lo16)
               (int)(2048u << 16), // tensor_dim0 hi=0 | tensor_dim1=2048 (lo16)
               (int)(512u << 16),  // tensor_dim1 hi=0 | tile_dim0=512
               64,                 // tile_dim1=64 | tile_dim2=0
               512,                // tensor_dim0_stride = 512 units (lo32)
               0,                  // stride0 hi16 | tensor_dim1_stride lo16
               0 };
    v4i g2 = {0, 0, 0, 0};         // 2D tile: dims 2..4 unused
    v4i g3 = {0, 0, 0, 0};
    v8i g4 = {0, 0, 0, 0, 0, 0, 0, 0};
    __builtin_amdgcn_tensor_load_to_lds(g0, g1, g2, g3, g4, 0);
    __builtin_amdgcn_s_wait_tensorcnt(0);
  }
#else
  {
    const uint4* src = (const uint4*)(w16 + (size_t)n_base * Hh);
    uint4* dst = (uint4*)wlds;
    for (int i = tid; i < 64 * 2048 / 8; i += 256) dst[i] = src[i];
  }
#endif
  __syncthreads();

  // 4x4 = 16 output tiles of 16x16; 8 waves -> 2 tiles each (same m, 2 n).
  const int mt  = wv >> 1;               // m-tile 0..3  (batch rows)
  const int ntp = (wv & 1) * 2;          // n-tile pair base 0 or 2

  _Float16* hb[2] = {hbuf0, hbuf1};

  const int nc0 = n_base + (ntp + 0) * 16 + l15;
  const int nc1 = n_base + (ntp + 1) * 16 + l15;
  const float bb0 = bhh[nc0], bb1 = bhh[nc1];
  const _Float16* brow0 = &wlds[((ntp + 0) * 16 + l15) * Hh];      // LDS
  const _Float16* brow1 = &wlds[((ntp + 1) * 16 + l15) * Hh];      // LDS

  for (int t = 0; t < Tt; ++t) {
    const _Float16* hc = hb[t & 1];
    _Float16* hn = hb[(t + 1) & 1];

    v8f c0 = {}, c1 = {};
    const _Float16* arow = hc + (size_t)(mt * 16 + l15) * Hh;      // global

    // Warm the xp (x_proj) lines we'll read in the epilogue while the
    // WMMA k-loop runs (global_prefetch_b8).
    {
      int b0 = mt * 16 + 8 * kh;
      __builtin_prefetch(&Out[((size_t)b0 * Tt + t) * Hh + nc0], 0, 1);
      __builtin_prefetch(&Out[((size_t)(b0 + 4) * Tt + t) * Hh + nc0], 0, 1);
    }

    for (int kb = 0; kb < Hh; kb += 32) {
      v16h af = cat8(*(const v8h*)(arow + kb + kh * 16),
                     *(const v8h*)(arow + kb + kh * 16 + 8));
      v16h b0 = cat8(*(const v8h*)(brow0 + kb + kh * 16),
                     *(const v8h*)(brow0 + kb + kh * 16 + 8));
      v16h b1 = cat8(*(const v8h*)(brow1 + kb + kh * 16),
                     *(const v8h*)(brow1 + kb + kh * 16 + 8));
      c0 = wmma_f16(af, b0, c0);
      c1 = wmma_f16(af, b1, c1);
    }

    // Fused epilogue: h_new = tanh(xp + noise[b,t] + C + b_hh[n]).
    #pragma unroll
    for (int r = 0; r < 8; ++r) {
      int b = mt * 16 + r + 8 * kh;                 // batch row 0..63
      float nz = noise[(size_t)b * Tt + t];
      size_t rowo = ((size_t)b * Tt + t) * (size_t)Hh;
      float v0 = tanhf(Out[rowo + nc0] + c0[r] + nz + bb0);
      float v1 = tanhf(Out[rowo + nc1] + c1[r] + nz + bb1);
      Out[rowo + nc0] = v0;
      Out[rowo + nc1] = v1;
      hn[(size_t)b * Hh + nc0] = (_Float16)v0;
      hn[(size_t)b * Hh + nc1] = (_Float16)v1;
    }

    // ---- grid-wide barrier (release/acquire at device scope) ----
    __threadfence();                 // each wave flushes its own h stores
    __syncthreads();
    if (tid == 0) {
      __hip_atomic_fetch_add(bar, 1u, __ATOMIC_ACQ_REL,
                             __HIP_MEMORY_SCOPE_AGENT);
      const unsigned target = (unsigned)NWG_REC * (unsigned)(t + 1);
      while (__hip_atomic_load(bar, __ATOMIC_ACQUIRE,
                               __HIP_MEMORY_SCOPE_AGENT) < target) {
        __builtin_amdgcn_s_sleep(2);
      }
    }
    __syncthreads();
    __threadfence();                 // acquire: invalidate before reading h
  }

  // Final state: steps ended with write into hb[(511+1)&1] == hbuf0.
  float* hlast = Out + (size_t)Bb * Tt * Hh;
  for (int i = tid; i < Bb * 64; i += 256) {
    int b = i >> 6;
    int col = n_base + (i & 63);
    hlast[(size_t)b * Hh + col] = (float)hbuf0[(size_t)b * Hh + col];
  }
}

// ---------------------------------------------------------------------------
extern "C" void kernel_launch(void* const* d_in, const int* in_sizes, int n_in,
                              void* d_out, int out_size, void* d_ws, size_t ws_size,
                              hipStream_t stream) {
  const float* input    = (const float*)d_in[0];   // [B,T,I]
  const float* internal = (const float*)d_in[1];   // [B,T]
  const float* w_ih     = (const float*)d_in[2];   // [H,I]
  const float* w_hh     = (const float*)d_in[3];   // [H,H]
  const float* b_ih     = (const float*)d_in[4];   // [H]
  const float* b_hh     = (const float*)d_in[5];   // [H]
  const float* h0       = (const float*)d_in[6];   // [1,B,H]
  // d_in[7] = trunc (stop_gradient boundary) — forward no-op.
  float* out = (float*)d_out;                      // [B,T,H] ++ [1,B,H]

  // Workspace layout (~8.6 MB): w_hh f16 | h ping | h pong | barrier word.
  char* ws = (char*)d_ws;
  _Float16* w16   = (_Float16*)ws;
  _Float16* hbuf0 = (_Float16*)(ws + (size_t)Hh * Hh * sizeof(_Float16));
  _Float16* hbuf1 = hbuf0 + (size_t)Bb * Hh;
  unsigned* bar   = (unsigned*)((char*)(hbuf1 + (size_t)Bb * Hh));

  prep_kernel<<<(Hh * Hh) / 256, 256, 0, stream>>>(w_hh, h0, w16, hbuf0, bar);

  dim3 g1(Hh / 128, BT / 128);   // (16, 256) macro-tiles
  xproj_gemm<<<g1, 256, 0, stream>>>(input, w_ih, b_ih, out);

  rnn_recur<<<NWG_REC, 256, 0, stream>>>(w16, internal, b_hh, out,
                                         hbuf0, hbuf1, bar);
}
